// TShape_model_84104049590450
// MI455X (gfx1250) — compile-verified, matrix operand order checked by reading, not run on hardware
//
#include <hip/hip_runtime.h>
#include <hip/hip_bf16.h>
#include <math.h>

typedef __attribute__((ext_vector_type(16))) _Float16 v16h;
typedef __attribute__((ext_vector_type(8)))  _Float16 v8h;
typedef __attribute__((ext_vector_type(8)))  float    v8f;

#define EPSF 1e-5f
// B=64, SEQ=8192, PATCH=2048, C=512, NPATCH=4 -> M = 64*4 = 256 rows

__device__ __forceinline__ float gelu_exact(float x) {
  return 0.5f * x * (1.0f + erff(x * 0.70710678118654752f));
}

// ---------------- f32 -> f16 convert (weights + activations) ----------------
__global__ void k_cvt16(const float* __restrict__ src, _Float16* __restrict__ dst, int n) {
  int i = blockIdx.x * blockDim.x + threadIdx.x;
  if (i < n) dst[i] = (_Float16)src[i];
}

// ------------- patch conv collapsed: sum + endpoints -> feat (256x512) -------
__global__ void k_patch(const float* __restrict__ x, const float* __restrict__ conv_w,
                        const float* __restrict__ conv_b, float* __restrict__ feat) {
  int bp  = blockIdx.x;     // 0..255  (b*4+p)
  int tid = threadIdx.x;    // 0..255
  const float* xp = x + (size_t)bp * 2048;
  float s = 0.f;
  for (int i = tid; i < 2048; i += 256) s += xp[i];
  __shared__ float red[256];
  __shared__ float sS, sx0, sxl;
  red[tid] = s;
  __syncthreads();
  for (int st = 128; st > 0; st >>= 1) {
    if (tid < st) red[tid] += red[tid + st];
    __syncthreads();
  }
  if (tid == 0) { sS = red[0]; sx0 = xp[0]; sxl = xp[2047]; }
  __syncthreads();
  float S = sS, x0 = sx0, xl = sxl;
  const float inv = 1.0f / 2048.0f;
  for (int c = tid; c < 512; c += 256) {
    float w0 = conv_w[c*3+0], w1 = conv_w[c*3+1], w2 = conv_w[c*3+2];
    feat[bp*512 + c] = ((S - xl)*w0 + S*w1 + (S - x0)*w2) * inv + conv_b[c];
  }
}

// -------- BatchNorm(train stats over 256 samples) + GELU + pos_embed --------
__global__ void k_bn(const float* __restrict__ feat, const float* __restrict__ bg,
                     const float* __restrict__ bb, const float* __restrict__ pos,
                     float* __restrict__ f, _Float16* __restrict__ fA) {
  int c = blockIdx.x * blockDim.x + threadIdx.x;   // 0..511 channel
  float s = 0.f, s2 = 0.f;
  for (int r = 0; r < 256; ++r) { float v = feat[r*512 + c]; s += v; s2 += v*v; }
  float mu  = s  * (1.f/256.f);
  float var = s2 * (1.f/256.f) - mu*mu;
  float sc = bg[c] * rsqrtf(var + EPSF);
  float sh = bb[c] - mu * sc;
  for (int r = 0; r < 256; ++r) {
    float v = feat[r*512 + c]*sc + sh;
    v = gelu_exact(v);
    v += pos[(r & 3)*512 + c];
    f[r*512 + c] = v;
    fA[r*512 + c] = (_Float16)v;
  }
}

// ---- local attention: seq=C=512, dim=4, nh=4 (head_dim=1) + LN over dim4 ----
__global__ void k_local(const float* __restrict__ f, const float* __restrict__ in_w,
                        const float* __restrict__ in_b, const float* __restrict__ out_w,
                        const float* __restrict__ out_b, const float* __restrict__ ln_g,
                        const float* __restrict__ ln_b, float* __restrict__ local_,
                        _Float16* __restrict__ comb) {
  int bb = blockIdx.x;    // batch
  int i  = threadIdx.x;   // position 0..511 (channel idx of original f)
  __shared__ float sk[4][512];
  __shared__ float sv[4][512];
  float xv[4];
  for (int d = 0; d < 4; ++d) xv[d] = f[(bb*4 + d)*512 + i];
  float q[4];
  for (int h = 0; h < 4; ++h) {
    float qq = in_b[h], kk = in_b[4+h], vv = in_b[8+h];
    for (int d = 0; d < 4; ++d) {
      qq += xv[d] * in_w[h*4 + d];
      kk += xv[d] * in_w[(4+h)*4 + d];
      vv += xv[d] * in_w[(8+h)*4 + d];
    }
    q[h] = qq; sk[h][i] = kk; sv[h][i] = vv;
  }
  __syncthreads();
  float oh[4];
  for (int h = 0; h < 4; ++h) {
    float m = -INFINITY, l = 0.f, acc = 0.f;
    for (int j = 0; j < 512; ++j) {           // head_dim=1 -> scale 1
      float sij = q[h] * sk[h][j];
      float mn = fmaxf(m, sij);
      float corr = expf(m - mn);
      float e = expf(sij - mn);
      l = l*corr + e;
      acc = acc*corr + e*sv[h][j];
      m = mn;
    }
    oh[h] = acc / l;
  }
  float o[4]; float mu = 0.f;
  for (int d = 0; d < 4; ++d) {
    float t = out_b[d];
    for (int h = 0; h < 4; ++h) t += oh[h] * out_w[d*4 + h];
    o[d] = t + xv[d];
    mu += o[d];
  }
  mu *= 0.25f;
  float var = 0.f;
  for (int d = 0; d < 4; ++d) { float dd = o[d]-mu; var += dd*dd; }
  var *= 0.25f;
  float inv = rsqrtf(var + EPSF);
  for (int d = 0; d < 4; ++d) {
    float val = (o[d]-mu)*inv*ln_g[d] + ln_b[d];
    local_[(bb*4 + d)*512 + i] = val;                 // transposed back (B,4,C)
    comb[(size_t)(bb*4 + d)*1024 + i] = (_Float16)val; // first half of concat
  }
}

// --------- generic WMMA GEMM: Out(MxN) = A(MxK,f16) @ W(NxK,f16)^T + bias ----
__global__ void k_gemm(const _Float16* __restrict__ A, const _Float16* __restrict__ Bw,
                       const float* __restrict__ bias, float* __restrict__ Out,
                       int M, int N, int K) {
  int tn = blockIdx.x, tm = blockIdx.y;
  int lane = threadIdx.x;          // one wave32 per 16x16 tile
  int hi  = lane >> 4;             // 0/1 selects K sub-block per ISA A layout
  int r15 = lane & 15;
  const _Float16* arow = A  + (size_t)(tm*16 + r15) * K + hi*8;
  const _Float16* brow = Bw + (size_t)(tn*16 + r15) * K + hi*8;
  v8f acc = {};
  for (int k = 0; k < K; k += 32) {
    v8h a_lo = *(const v8h*)(arow + k);
    v8h a_hi = *(const v8h*)(arow + k + 16);
    v8h b_lo = *(const v8h*)(brow + k);
    v8h b_hi = *(const v8h*)(brow + k + 16);
    v16h a, b;
    #pragma unroll
    for (int t = 0; t < 8; ++t) { a[t] = a_lo[t]; a[t+8] = a_hi[t];
                                  b[t] = b_lo[t]; b[t+8] = b_hi[t]; }
    acc = __builtin_amdgcn_wmma_f32_16x16x32_f16(false, a, false, b,
                                                 (short)0, acc, false, false);
  }
  int col = tn*16 + r15;
  float bv = bias ? bias[col] : 0.0f;
  #pragma unroll
  for (int r = 0; r < 8; ++r) {
    int row = tm*16 + r + hi*8;
    Out[(size_t)row*N + col] = acc[r] + bv;
  }
}

// -------- global attention: seq=4, dim=512, nh=4, head_dim=128 --------------
__global__ void k_gattn(const float* __restrict__ qkv, _Float16* __restrict__ attnA) {
  int b = blockIdx.x >> 2;
  int h = blockIdx.x & 3;
  int d = threadIdx.x;             // 0..127
  float q[4], k[4], v[4];
  for (int i = 0; i < 4; ++i) {
    const float* row = qkv + (size_t)(b*4 + i)*1536 + h*128 + d;
    q[i] = row[0];
    k[i] = row[512];
    v[i] = row[1024];
  }
  __shared__ float part[16][128];
  __shared__ float attn[4][4];
  for (int i = 0; i < 4; ++i)
    for (int j = 0; j < 4; ++j)
      part[i*4 + j][d] = q[i]*k[j];
  __syncthreads();
  if (d < 16) {
    float s = 0.f;
    for (int t = 0; t < 128; ++t) s += part[d][t];
    part[d][0] = s;
  }
  __syncthreads();
  if (d < 4) {
    const float scale = 0.0883883476483184f;   // 1/sqrt(128)
    float sc[4], mx = -INFINITY;
    for (int j = 0; j < 4; ++j) { sc[j] = part[d*4 + j][0]*scale; mx = fmaxf(mx, sc[j]); }
    float sum = 0.f;
    for (int j = 0; j < 4; ++j) { sc[j] = expf(sc[j]-mx); sum += sc[j]; }
    for (int j = 0; j < 4; ++j) attn[d][j] = sc[j]/sum;
  }
  __syncthreads();
  for (int i = 0; i < 4; ++i) {
    float o = 0.f;
    for (int j = 0; j < 4; ++j) o += attn[i][j]*v[j];
    attnA[(size_t)(b*4 + i)*512 + h*128 + d] = (_Float16)o;
  }
}

// ---------- g = LN(g_pre + f) over dim 512; also second half of concat ------
__global__ void k_lng(const float* __restrict__ g_pre, const float* __restrict__ f,
                      const float* __restrict__ ln_g, const float* __restrict__ ln_b,
                      float* __restrict__ g, _Float16* __restrict__ comb) {
  int r = blockIdx.x, t = threadIdx.x;   // 256 threads / 512 elems
  float v0 = g_pre[r*512 + t]       + f[r*512 + t];
  float v1 = g_pre[r*512 + t + 256] + f[r*512 + t + 256];
  __shared__ float rs[256], rs2[256];
  rs[t] = v0 + v1; rs2[t] = v0*v0 + v1*v1;
  __syncthreads();
  for (int st = 128; st > 0; st >>= 1) {
    if (t < st) { rs[t] += rs[t+st]; rs2[t] += rs2[t+st]; }
    __syncthreads();
  }
  float mu  = rs[0]  * (1.f/512.f);
  float var = rs2[0] * (1.f/512.f) - mu*mu;
  float inv = rsqrtf(var + EPSF);
  float o0 = (v0-mu)*inv*ln_g[t]     + ln_b[t];
  float o1 = (v1-mu)*inv*ln_g[t+256] + ln_b[t+256];
  g[r*512 + t] = o0; g[r*512 + t + 256] = o1;
  comb[(size_t)r*1024 + 512 + t]       = (_Float16)o0;
  comb[(size_t)r*1024 + 512 + t + 256] = (_Float16)o1;
}

// ------------- fused = sigmoid(gate_pre)*local + (1-s)*g, to f16 ------------
__global__ void k_fuse(const float* __restrict__ gate_pre, const float* __restrict__ local_,
                       const float* __restrict__ g, _Float16* __restrict__ fusedA) {
  int i = blockIdx.x*blockDim.x + threadIdx.x;   // 0..131071
  float gt = 1.0f / (1.0f + expf(-gate_pre[i]));
  fusedA[i] = (_Float16)(gt*local_[i] + (1.0f-gt)*g[i]);
}

// ------- tr = LN(gelu(tr_pre)) + residual(f); row_agg = tr . agg_w + b ------
__global__ void k_trln(const float* __restrict__ tr_pre, const float* __restrict__ f,
                       const float* __restrict__ ln_g, const float* __restrict__ ln_b,
                       const float* __restrict__ agg_w, const float* __restrict__ agg_b,
                       float* __restrict__ row_agg) {
  int r = blockIdx.x, t = threadIdx.x;
  float u0 = gelu_exact(tr_pre[r*512 + t]);
  float u1 = gelu_exact(tr_pre[r*512 + t + 256]);
  __shared__ float rs[256], rs2[256];
  rs[t] = u0 + u1; rs2[t] = u0*u0 + u1*u1;
  __syncthreads();
  for (int st = 128; st > 0; st >>= 1) {
    if (t < st) { rs[t] += rs[t+st]; rs2[t] += rs2[t+st]; }
    __syncthreads();
  }
  float mu  = rs[0]  * (1.f/512.f);
  float var = rs2[0] * (1.f/512.f) - mu*mu;
  float inv = rsqrtf(var + EPSF);
  float o0 = (u0-mu)*inv*ln_g[t]     + ln_b[t]     + f[r*512 + t];
  float o1 = (u1-mu)*inv*ln_g[t+256] + ln_b[t+256] + f[r*512 + t + 256];
  float dot = o0*agg_w[t] + o1*agg_w[t+256];
  __syncthreads();
  rs[t] = dot;
  __syncthreads();
  for (int st = 128; st > 0; st >>= 1) {
    if (t < st) rs[t] += rs[t+st];
    __syncthreads();
  }
  if (t == 0) row_agg[r] = rs[0] + agg_b[0];
}

// ---------------- out[b] = sum_p row_agg[b,p]*seq_w[p] + seq_b --------------
__global__ void k_final(const float* __restrict__ row_agg, const float* __restrict__ seq_w,
                        const float* __restrict__ seq_b, float* __restrict__ out) {
  int b = threadIdx.x;
  if (b < 64) {
    float s = seq_b[0];
    for (int p = 0; p < 4; ++p) s += row_agg[b*4 + p]*seq_w[p];
    out[b] = s;
  }
}

extern "C" void kernel_launch(void* const* d_in, const int* in_sizes, int n_in,
                              void* d_out, int out_size, void* d_ws, size_t ws_size,
                              hipStream_t stream) {
  const float* x        = (const float*)d_in[0];
  const float* conv_w   = (const float*)d_in[1];
  const float* conv_b   = (const float*)d_in[2];
  const float* bn_gamma = (const float*)d_in[3];
  const float* bn_beta  = (const float*)d_in[4];
  const float* pos      = (const float*)d_in[5];
  const float* l_in_w   = (const float*)d_in[6];
  const float* l_in_b   = (const float*)d_in[7];
  const float* l_out_w  = (const float*)d_in[8];
  const float* l_out_b  = (const float*)d_in[9];
  const float* ln2_g    = (const float*)d_in[10];
  const float* ln2_b    = (const float*)d_in[11];
  const float* g_in_w   = (const float*)d_in[12];
  const float* g_in_b   = (const float*)d_in[13];
  const float* g_out_w  = (const float*)d_in[14];
  const float* g_out_b  = (const float*)d_in[15];
  const float* ln1_g    = (const float*)d_in[16];
  const float* ln1_b    = (const float*)d_in[17];
  const float* gate_w   = (const float*)d_in[18];
  const float* gate_b   = (const float*)d_in[19];
  const float* tr_w     = (const float*)d_in[20];
  const float* tr_b     = (const float*)d_in[21];
  const float* tr_ln_g  = (const float*)d_in[22];
  const float* tr_ln_b  = (const float*)d_in[23];
  const float* agg_w    = (const float*)d_in[24];
  const float* agg_b    = (const float*)d_in[25];
  const float* seq_w    = (const float*)d_in[26];
  const float* seq_b    = (const float*)d_in[27];
  float* out = (float*)d_out;

  const int M = 256, C = 512;
  size_t off = 0;
  auto wsalloc = [&](size_t bytes) {
    void* p = (char*)d_ws + off;
    off += (bytes + 255) & ~(size_t)255;
    return p;
  };
  float*    feat    = (float*)wsalloc((size_t)M*C*4);
  float*    f       = (float*)wsalloc((size_t)M*C*4);
  float*    qkv     = (float*)wsalloc((size_t)M*3*C*4);
  float*    local_  = (float*)wsalloc((size_t)M*C*4);
  float*    g_pre   = (float*)wsalloc((size_t)M*C*4);
  float*    gbuf    = (float*)wsalloc((size_t)M*C*4);
  float*    gatep   = (float*)wsalloc((size_t)M*C*4);
  float*    trp     = (float*)wsalloc((size_t)M*C*4);
  float*    row_agg = (float*)wsalloc((size_t)M*4);
  _Float16* fA      = (_Float16*)wsalloc((size_t)M*C*2);
  _Float16* attnA   = (_Float16*)wsalloc((size_t)M*C*2);
  _Float16* comb    = (_Float16*)wsalloc((size_t)M*2*C*2);
  _Float16* fusedA  = (_Float16*)wsalloc((size_t)M*C*2);
  _Float16* w_gin   = (_Float16*)wsalloc((size_t)3*C*C*2);
  _Float16* w_gout  = (_Float16*)wsalloc((size_t)C*C*2);
  _Float16* w_gate  = (_Float16*)wsalloc((size_t)C*2*C*2);
  _Float16* w_tr    = (_Float16*)wsalloc((size_t)C*C*2);

  // weight f16 conversion (cheap; redone every call for determinism)
  k_cvt16<<<(3*C*C + 255)/256, 256, 0, stream>>>(g_in_w,  w_gin,  3*C*C);
  k_cvt16<<<(C*C   + 255)/256, 256, 0, stream>>>(g_out_w, w_gout, C*C);
  k_cvt16<<<(2*C*C + 255)/256, 256, 0, stream>>>(gate_w,  w_gate, 2*C*C);
  k_cvt16<<<(C*C   + 255)/256, 256, 0, stream>>>(tr_w,    w_tr,   C*C);

  // collapsed patch conv -> feat
  k_patch<<<256, 256, 0, stream>>>(x, conv_w, conv_b, feat);
  // BN(train) + GELU + pos_embed -> f (f32 + f16)
  k_bn<<<2, 256, 0, stream>>>(feat, bn_gamma, bn_beta, pos, f, fA);
  // local attention (dim=4) + LN2 -> local (f32) + first half of concat (f16)
  k_local<<<64, 512, 0, stream>>>(f, l_in_w, l_in_b, l_out_w, l_out_b,
                                  ln2_g, ln2_b, local_, comb);
  // qkv = f @ g_in_w^T + b   (256 x 1536, K=512) via WMMA
  k_gemm<<<dim3(96, 16), 32, 0, stream>>>(fA, w_gin, g_in_b, qkv, M, 3*C, C);
  // global attention (seq=4, head_dim=128) -> attnA f16
  k_gattn<<<256, 128, 0, stream>>>(qkv, attnA);
  // out proj (256 x 512, K=512)
  k_gemm<<<dim3(32, 16), 32, 0, stream>>>(attnA, w_gout, g_out_b, g_pre, M, C, C);
  // g = LN1(g_pre + f) -> g (f32) + second half of concat (f16)
  k_lng<<<256, 256, 0, stream>>>(g_pre, f, ln1_g, ln1_b, gbuf, comb);
  // gate_pre = concat @ gate_w^T + b  (256 x 512, K=1024)
  k_gemm<<<dim3(32, 16), 32, 0, stream>>>(comb, w_gate, gate_b, gatep, M, C, 2*C);
  // fused = sigmoid(gate)*local + (1-sigmoid)*g  -> f16
  k_fuse<<<(M*C + 255)/256, 256, 0, stream>>>(gatep, local_, gbuf, fusedA);
  // tr_pre = fused @ tr_w^T + b  (256 x 512, K=512)
  k_gemm<<<dim3(32, 16), 32, 0, stream>>>(fusedA, w_tr, tr_b, trp, M, C, C);
  // tr = LN(gelu(tr_pre)) + residual; row dot with agg_w
  k_trln<<<256, 256, 0, stream>>>(trp, f, tr_ln_g, tr_ln_b, agg_w, agg_b, row_agg);
  // final (B,4) @ seq_w^T + seq_b
  k_final<<<1, 64, 0, stream>>>(row_agg, seq_w, seq_b, out);
}
